// RegLSTM_28965259444708
// MI455X (gfx1250) — matrix operand order repaired and checked by reading, not compiled, not run on hardware
//
#include <hip/hip_runtime.h>
#include <hip/hip_bf16.h>
#include <stdint.h>

#define B_ 32
#define T_ 512
#define I_ 512
#define H_ 512
#define G_ 2048  /* 4*H */

typedef __bf16 bf16_t;
typedef __attribute__((ext_vector_type(16))) __bf16 v16bf;
typedef __attribute__((ext_vector_type(8)))  __bf16 v8bf;
typedef __attribute__((ext_vector_type(8)))  float  v8f;
typedef __attribute__((ext_vector_type(4)))  unsigned int u32x4;
typedef __attribute__((ext_vector_type(8)))  int i32x8;
typedef __attribute__((ext_vector_type(4)))  int i32x4;

// AS-qualified pointer element types for the async-LDS builtin
typedef __attribute__((address_space(1))) i32x4 gas_i32x4;   // global
typedef __attribute__((address_space(3))) i32x4 las_i32x4;   // LDS

// ---- CDNA5 feature detection -------------------------------------------------
#if __has_builtin(__builtin_amdgcn_tensor_load_to_lds)
#define HAVE_TDM 1
#else
#define HAVE_TDM 0
#endif

#if __has_builtin(__builtin_amdgcn_global_load_async_to_lds_b128)
#define HAVE_ASYNC_LDS 1
#else
#define HAVE_ASYNC_LDS 0
#endif

__device__ __forceinline__ void async_copy16(void* lds_dst, const void* gsrc) {
#if HAVE_ASYNC_LDS
    __builtin_amdgcn_global_load_async_to_lds_b128(
        (gas_i32x4*)gsrc, (las_i32x4*)lds_dst, 0, 0);
#else
    asm volatile("global_load_async_to_lds_b128 %0, %1, off"
                 :: "v"((uint32_t)(uintptr_t)lds_dst),
                    "v"((uint64_t)(uintptr_t)gsrc)
                 : "memory");
#endif
}

#if __has_builtin(__builtin_amdgcn_s_wait_asynccnt)
#define WAIT_ASYNC0() __builtin_amdgcn_s_wait_asynccnt(0)
#define WAIT_ASYNC1() __builtin_amdgcn_s_wait_asynccnt(1)
#else
#define WAIT_ASYNC0() asm volatile("s_wait_asynccnt 0x0" ::: "memory")
#define WAIT_ASYNC1() asm volatile("s_wait_asynccnt 0x1" ::: "memory")
#endif

__device__ __forceinline__ float sigmoidf_(float x) {
    return 1.0f / (1.0f + __expf(-x));
}

// stretched-concrete hard gate: z in [0,1]
__device__ __forceinline__ float gate_z(float e, float m) {
    float lg = __logf(e) - log1pf(-e);
    float y  = sigmoidf_((lg + m) * 1.5f);     // /TEMP, TEMP = 2/3
    float z  = y * 1.2f - 0.1f;                // *(LIM_B-LIM_A) + LIM_A
    return fminf(fmaxf(z, 0.0f), 1.0f);
}

// ---------------- elementwise gating: out[b,w] = z(eps[b,w], mask[w]) * param[w] (bf16)
__global__ __launch_bounds__(256) void gate_weights_kernel(
    const float* __restrict__ eps, const float* __restrict__ mask,
    const float* __restrict__ param, bf16_t* __restrict__ out,
    long long n, int per)
{
    long long idx = (long long)blockIdx.x * 256 + threadIdx.x;
    if (idx >= n) return;
    int widx = (int)(idx % per);
    float z = gate_z(eps[idx], mask[widx]);
    out[idx] = (bf16_t)(z * param[widx]);
}

// ---------------- gated bias: bias[b,g] = z_ih*b_ih[g] + z_hh*b_hh[g]
__global__ __launch_bounds__(256) void gate_bias_kernel(
    const float* __restrict__ eps_bih, const float* __restrict__ eps_bhh,
    const float* __restrict__ mask_bih, const float* __restrict__ mask_bhh,
    const float* __restrict__ b_ih, const float* __restrict__ b_hh,
    float* __restrict__ bias, int n)
{
    int idx = blockIdx.x * 256 + threadIdx.x;
    if (idx >= n) return;
    int g = idx % G_;
    float z1 = gate_z(eps_bih[idx], mask_bih[g]);
    float z2 = gate_z(eps_bhh[idx], mask_bhh[g]);
    bias[idx] = z1 * b_ih[g] + z2 * b_hh[g];
}

// ---------------- fp32 -> bf16 convert (for seq)
__global__ __launch_bounds__(256) void f32_to_bf16_kernel(
    const float* __restrict__ in, bf16_t* __restrict__ out, long long n)
{
    long long idx = (long long)blockIdx.x * 256 + threadIdx.x;
    if (idx < n) out[idx] = (bf16_t)in[idx];
}

// ---------------- x_proj[b,t,g] = seq[b,t,:] . Wih_g[b,g,:] + bias[b,g]
// A tile (16 x 512 bf16, shared by all 8 waves) staged in LDS by the Tensor Data
// Mover (with pad_interval=256dw / pad_amount=4dw -> 1040B row stride, bank-spread),
// then per-wave 16x16 C tiles via v_wmma_f32_16x16x32_bf16 over K.
#define XROW_BYTES 1040  /* 512*2 + 16B TDM pad per row */

__global__ __launch_bounds__(256) void xproj_wmma_kernel(
    const bf16_t* __restrict__ Xb16,   // [B][T][I]
    const bf16_t* __restrict__ Wih,    // [B][G][I]
    const float*  __restrict__ bias,   // [B][G]
    float* __restrict__ xp)            // [B][T][G]
{
    const int lane = threadIdx.x & 31;
    const int wave = threadIdx.x >> 5;
    const int half = lane >> 4;
    const int l15  = lane & 15;
    const int b  = blockIdx.z;
    const int t0 = blockIdx.y * 16;
    const int n0 = (blockIdx.x * 8 + wave) * 16;

    const bf16_t* Wrow = Wih + ((size_t)b * G_ + (n0 + l15)) * I_;

#if HAVE_TDM
    __shared__ __align__(16) char xtile[16 * XROW_BYTES];
    if (threadIdx.x < 32) {   // one wave issues one TDM descriptor
        uint64_t ga  = (uint64_t)(uintptr_t)(Xb16 + ((size_t)b * T_ + t0) * I_);
        uint32_t lds = (uint32_t)(uintptr_t)&xtile[0];
        // D# group0: count=1, lds_addr, 57b global_addr, type=2
        u32x4 g0 = { 1u, lds, (uint32_t)ga, (uint32_t)(ga >> 32) | (2u << 30) };
        // D# group1: data_size=2B, pad_enable, pad_interval=256dw, pad_amount=4dw,
        // tensor_dim0=512, tensor_dim1=16, tile_dim0=512, tile_dim1=16, dim0_stride=512
        i32x8 g1 = { (int)0x07D10000,      // mask=0 | ds=1<<16 | pad_en | intv=7 | amt=3
                     (int)(512u << 16),    // abar=0, tensor_dim0[15:0]=512
                     (int)(16u << 16),     // tensor_dim0[31:16]=0, tensor_dim1[15:0]=16
                     (int)(512u << 16),    // tensor_dim1[31:16]=0, tile_dim0=512
                     16,                   // tile_dim1=16, tile_dim2=0
                     512, 0, 0 };          // tensor_dim0_stride=512, dim1_stride=0
        i32x4 gz = { 0, 0, 0, 0 };
#if __clang_major__ >= 23
        i32x8 gz8 = { 0, 0, 0, 0, 0, 0, 0, 0 };
        __builtin_amdgcn_tensor_load_to_lds(g0, g1, gz, gz, gz8, 0);
#else
        __builtin_amdgcn_tensor_load_to_lds(g0, g1, gz, gz, 0);
#endif
    }
    __builtin_amdgcn_s_wait_tensorcnt(0);
    __syncthreads();
    const char* xrow = xtile + l15 * XROW_BYTES;
#else
    const bf16_t* Xrow = Xb16 + ((size_t)b * T_ + (t0 + l15)) * I_;
#endif

    v8f acc = {};
    for (int k0 = 0; k0 < I_; k0 += 32) {
        // A fragment: element e holds k = k0 + (e>>3)*16 + half*8 + (e&7), row = t0+l15
#if HAVE_TDM
        v8bf a0 = *(const v8bf*)(xrow + (k0 + half * 8) * 2);
        v8bf a1 = *(const v8bf*)(xrow + (k0 + 16 + half * 8) * 2);
#else
        v8bf a0 = *(const v8bf*)(Xrow + k0 + half * 8);
        v8bf a1 = *(const v8bf*)(Xrow + k0 + 16 + half * 8);
#endif
        // B fragment: element e holds k = k0 + (e>>3)*16 + half*8 + (e&7), col = n0+l15
        v8bf b0 = *(const v8bf*)(Wrow + k0 + half * 8);
        v8bf b1 = *(const v8bf*)(Wrow + k0 + 16 + half * 8);
        v16bf A, Bm;
#pragma unroll
        for (int e = 0; e < 8; ++e) {
            A[e] = a0[e];  A[8 + e] = a1[e];
            Bm[e] = b0[e]; Bm[8 + e] = b1[e];
        }
        acc = __builtin_amdgcn_wmma_f32_16x16x32_bf16(
            /*neg_a=*/false, A, /*neg_b=*/false, Bm,
            /*c_mod=*/(short)0, acc, /*reuse_a=*/false, /*reuse_b=*/false);
    }
    // C/D layout: VGPR r, lanes 0-15 -> M=r, lanes 16-31 -> M=r+8; N = lane&15
    float bv = bias[(size_t)b * G_ + n0 + l15];
#pragma unroll
    for (int r = 0; r < 8; ++r) {
        int m = t0 + r + half * 8;
        xp[((size_t)b * T_ + m) * G_ + (n0 + l15)] = acc[r] + bv;
    }
}

// ---------------- sequential LSTM recurrence: one 1024-thread workgroup per batch.
// gates[g] = xp_lds[t][g] + sum_k Whh_g[b,g,k] * h[k]   (Whh bf16, L2-resident).
// xp rows are double-buffered in LDS via async global->LDS loads: the row for
// step t+1 streams in (ASYNCcnt) while step t computes.
__global__ __launch_bounds__(1024) void lstm_recurrence_kernel(
    const float*  __restrict__ xp,     // [B][T][G]
    const bf16_t* __restrict__ Whh,    // [B][G][H]
    float* __restrict__ out)           // [B][T][H] ++ h_n[B][H] ++ c_n[B][H]
{
    __shared__ __align__(16) float h_sh[H_];
    __shared__ float gates_sh[G_];
    __shared__ __align__(16) float xp_lds[2][G_];
    const int b   = blockIdx.x;
    const int tid = threadIdx.x;

    const float* xp_b = xp + (size_t)b * T_ * G_;

    float c = 0.0f, h_last = 0.0f;
    if (tid < H_) h_sh[tid] = 0.0f;
    if (tid < 512) async_copy16(&xp_lds[0][tid * 4], xp_b + tid * 4);  // row t=0

    const float4* h4 = (const float4*)h_sh;

    for (int t = 0; t < T_; ++t) {
        // prefetch xp row t+1 into the other buffer, then wait for row t
        if (t + 1 < T_) {
            if (tid < 512)
                async_copy16(&xp_lds[(t + 1) & 1][tid * 4],
                             xp_b + (size_t)(t + 1) * G_ + tid * 4);
            WAIT_ASYNC1();   // in-order completion => row t resident
        } else {
            WAIT_ASYNC0();
        }
        __syncthreads();

        const float* xp_t = xp_lds[t & 1];
#pragma unroll
        for (int gg = 0; gg < 2; ++gg) {
            const int g = tid + gg * 1024;
            const uint4* wrow = (const uint4*)(Whh + ((size_t)b * G_ + g) * H_);
            float acc = xp_t[g];
#pragma unroll 4
            for (int k4 = 0; k4 < H_ / 8; ++k4) {
                uint4  w  = wrow[k4];              // 8 bf16 weights
                float4 ha = h4[2 * k4];
                float4 hb = h4[2 * k4 + 1];
                acc += __uint_as_float(w.x << 16)         * ha.x;
                acc += __uint_as_float(w.x & 0xffff0000u) * ha.y;
                acc += __uint_as_float(w.y << 16)         * ha.z;
                acc += __uint_as_float(w.y & 0xffff0000u) * ha.w;
                acc += __uint_as_float(w.z << 16)         * hb.x;
                acc += __uint_as_float(w.z & 0xffff0000u) * hb.y;
                acc += __uint_as_float(w.w << 16)         * hb.z;
                acc += __uint_as_float(w.w & 0xffff0000u) * hb.w;
            }
            gates_sh[g] = acc;
        }
        __syncthreads();
        if (tid < H_) {
            float i  = sigmoidf_(gates_sh[tid]);
            float f  = sigmoidf_(gates_sh[tid + H_]);
            float gt = tanhf(gates_sh[tid + 2 * H_]);
            float o  = sigmoidf_(gates_sh[tid + 3 * H_]);
            c = f * c + i * gt;
            float hn = o * tanhf(c);
            out[((size_t)b * T_ + t) * H_ + tid] = hn;
            h_sh[tid] = hn;
            h_last = hn;
        }
        __syncthreads();
    }
    if (tid < H_) {
        float* hn_out = out + (size_t)B_ * T_ * H_;
        float* cn_out = hn_out + (size_t)B_ * H_;
        hn_out[(size_t)b * H_ + tid] = h_last;
        cn_out[(size_t)b * H_ + tid] = c;
    }
}

extern "C" void kernel_launch(void* const* d_in, const int* in_sizes, int n_in,
                              void* d_out, int out_size, void* d_ws, size_t ws_size,
                              hipStream_t stream)
{
    const float* seq      = (const float*)d_in[0];
    const float* w_ih     = (const float*)d_in[1];
    const float* w_hh     = (const float*)d_in[2];
    const float* b_ih     = (const float*)d_in[3];
    const float* b_hh     = (const float*)d_in[4];
    const float* mask_ih  = (const float*)d_in[5];
    const float* mask_hh  = (const float*)d_in[6];
    const float* mask_bih = (const float*)d_in[7];
    const float* mask_bhh = (const float*)d_in[8];
    const float* eps_ih   = (const float*)d_in[9];
    const float* eps_hh   = (const float*)d_in[10];
    const float* eps_bih  = (const float*)d_in[11];
    const float* eps_bhh  = (const float*)d_in[12];
    float* out = (float*)d_out;

    // workspace layout (bytes)
    char* ws = (char*)d_ws;
    const size_t GIbytes  = (size_t)B_ * G_ * I_ * sizeof(bf16_t);  // 64 MB
    const size_t SEQbytes = (size_t)B_ * T_ * I_ * sizeof(bf16_t);  // 16 MB
    const size_t BGbytes  = (size_t)B_ * G_ * sizeof(float);        // 256 KB
    bf16_t* Wih_g = (bf16_t*)(ws);
    bf16_t* Whh_g = (bf16_t*)(ws + GIbytes);
    bf16_t* seq16 = (bf16_t*)(ws + 2 * GIbytes);
    float*  biasv = (float*) (ws + 2 * GIbytes + SEQbytes);
    float*  xpv   = (float*) (ws + 2 * GIbytes + SEQbytes + BGbytes);

    const long long nIH = (long long)B_ * G_ * I_;   // 33.5M
    const long long nHH = (long long)B_ * G_ * H_;   // 33.5M
    const long long nSQ = (long long)B_ * T_ * I_;   // 8.4M

    gate_weights_kernel<<<(unsigned)((nIH + 255) / 256), 256, 0, stream>>>(
        eps_ih, mask_ih, w_ih, Wih_g, nIH, G_ * I_);
    gate_weights_kernel<<<(unsigned)((nHH + 255) / 256), 256, 0, stream>>>(
        eps_hh, mask_hh, w_hh, Whh_g, nHH, G_ * H_);
    gate_bias_kernel<<<(B_ * G_ + 255) / 256, 256, 0, stream>>>(
        eps_bih, eps_bhh, mask_bih, mask_bhh, b_ih, b_hh, biasv, B_ * G_);
    f32_to_bf16_kernel<<<(unsigned)((nSQ + 255) / 256), 256, 0, stream>>>(
        seq, seq16, nSQ);

    dim3 gGrid(G_ / 128, T_ / 16, B_);   // 8 waves/block, 16 cols each
    xproj_wmma_kernel<<<gGrid, 256, 0, stream>>>(seq16, Wih_g, biasv, xpv);

    lstm_recurrence_kernel<<<B_, 1024, 0, stream>>>(xpv, Whh_g, out);
}
